// PointTransformerCls_54786602828256
// MI455X (gfx1250) — matrix operand order repaired and checked by reading, not compiled, not run on hardware
//
#include <hip/hip_runtime.h>
#include <cstddef>
#include <cstdint>

typedef _Float16 half_t;
typedef __attribute__((ext_vector_type(16))) _Float16 v16h;
typedef __attribute__((ext_vector_type(8)))  _Float16 v8h;
typedef __attribute__((ext_vector_type(8)))  float    v8f;

#define THREADS 256
static inline unsigned gblocks(long long n, int t = THREADS) {
  return (unsigned)((n + t - 1) / t);
}

// ---------------------------------------------------------------------------
// Weight convert: W (K,N) f32 row-major -> Wt (Np,Kp) f16, transposed + padded
// ---------------------------------------------------------------------------
__global__ void k_cvt_w(const float* __restrict__ W, half_t* __restrict__ Wt,
                        int K, int N, int Kp, int Np) {
  long long t = (long long)blockIdx.x * blockDim.x + threadIdx.x;
  long long tot = (long long)Np * Kp;
  if (t >= tot) return;
  int n = (int)(t / Kp);
  int k = (int)(t % Kp);
  Wt[t] = (n < N && k < K) ? (half_t)W[(long long)k * N + n] : (half_t)0.f;
}

// ---------------------------------------------------------------------------
// Activation convert: A (Bt,M,K) f32 -> At (Bt,Mp,Kp) f16 zero-padded.
// Removes all predication/conversion from the WMMA hot loop.
// ---------------------------------------------------------------------------
__global__ void k_cvt_a(const float* __restrict__ A, half_t* __restrict__ At,
                        int M, int K, int Mp, int Kp) {
  const int b = blockIdx.z;
  A  += (long long)b * M * K;
  At += (long long)b * Mp * Kp;
  long long t = (long long)blockIdx.x * blockDim.x + threadIdx.x;
  long long tot = (long long)Mp * Kp;
  if (t >= tot) return;
  int m = (int)(t / Kp);
  int k = (int)(t % Kp);
  At[t] = (m < M && k < K) ? (half_t)A[(long long)m * K + k] : (half_t)0.f;
}

// ---------------------------------------------------------------------------
// WMMA GEMM: C[b] = act( A[b] (M,K) @ W (K,N) + bias )
// At f16 (Mp x Kp, Kp%64==0), Wt f16 (Np x Kp, Np%32==0).
// One wave -> 16x32 C tile (2 accumulators share one A fragment);
// K-loop unrolled 2x -> 4 v_wmma per iteration, no predication inside.
// A 16x32 / B 32x16 use contiguous k = 16*hi + s in both; f32 C layout per CDNA5 ISA 7.12.2.
// ---------------------------------------------------------------------------
__global__ void k_gemm(const half_t* __restrict__ At, const half_t* __restrict__ Wt,
                       const float* __restrict__ bias, float* __restrict__ C,
                       int M, int Mp, int Kp, int N, int relu) {
  const int b = blockIdx.z;
  At += (long long)b * Mp * Kp;
  C  += (long long)b * M * N;
  const int wave = threadIdx.x >> 5;
  const int lane = threadIdx.x & 31;
  const int hi   = lane >> 4;          // half-wave select
  const int lrow = lane & 15;
  const int m0 = blockIdx.x * 16;
  const int n0 = (blockIdx.y * (blockDim.x >> 5) + wave) * 32;
  const int Np = (N + 31) & ~31;
  if (n0 >= Np) return;                // wave-uniform: EXEC all-ones for WMMA

  v8f acc0 = {0.f, 0.f, 0.f, 0.f, 0.f, 0.f, 0.f, 0.f};
  v8f acc1 = {0.f, 0.f, 0.f, 0.f, 0.f, 0.f, 0.f, 0.f};

  const half_t* Arow  = At + (long long)(m0 + lrow) * Kp;        // m0+lrow < Mp
  const half_t* Brow0 = Wt + (long long)(n0 + lrow) * Kp;
  const half_t* Brow1 = Wt + (long long)(n0 + 16 + lrow) * Kp;

  for (int kb = 0; kb < Kp; kb += 64) {
    __builtin_prefetch(Arow + kb + 64, 0, 3);   // global_prefetch_b8, near-cache
    // ---- k-step 0: K = kb .. kb+31 ----
    v8h a0lo = *(const v8h*)(Arow + kb + hi * 16);        // elems 0..7
    v8h a0hi = *(const v8h*)(Arow + kb + 8 + hi * 16);    // elems 8..15
    v16h a0 = __builtin_shufflevector(a0lo, a0hi,
        0, 1, 2, 3, 4, 5, 6, 7, 8, 9, 10, 11, 12, 13, 14, 15);
    v16h b0 = *(const v16h*)(Brow0 + kb + hi * 16);
    v16h b1 = *(const v16h*)(Brow1 + kb + hi * 16);
    acc0 = __builtin_amdgcn_wmma_f32_16x16x32_f16(false, a0, false, b0,
                                                  (short)0, acc0, false, false);
    acc1 = __builtin_amdgcn_wmma_f32_16x16x32_f16(false, a0, false, b1,
                                                  (short)0, acc1, false, false);
    // ---- k-step 1: K = kb+32 .. kb+63 ----
    v8h a1lo = *(const v8h*)(Arow + kb + 32 + hi * 16);
    v8h a1hi = *(const v8h*)(Arow + kb + 40 + hi * 16);
    v16h a1 = __builtin_shufflevector(a1lo, a1hi,
        0, 1, 2, 3, 4, 5, 6, 7, 8, 9, 10, 11, 12, 13, 14, 15);
    v16h b2 = *(const v16h*)(Brow0 + kb + 32 + hi * 16);
    v16h b3 = *(const v16h*)(Brow1 + kb + 32 + hi * 16);
    acc0 = __builtin_amdgcn_wmma_f32_16x16x32_f16(false, a1, false, b2,
                                                  (short)0, acc0, false, false);
    acc1 = __builtin_amdgcn_wmma_f32_16x16x32_f16(false, a1, false, b3,
                                                  (short)0, acc1, false, false);
  }

  const int col0 = n0 + lrow;
  const int col1 = n0 + 16 + lrow;
#pragma unroll
  for (int r = 0; r < 8; ++r) {
    int row = m0 + hi * 8 + r;
    if (row < M) {
      if (col0 < N) {
        float v = acc0[r];
        if (bias) v += bias[col0];
        if (relu) v = fmaxf(v, 0.f);
        C[(long long)row * N + col0] = v;
      }
      if (col1 < N) {
        float v = acc1[r];
        if (bias) v += bias[col1];
        if (relu) v = fmaxf(v, 0.f);
        C[(long long)row * N + col1] = v;
      }
    }
  }
}

// ---------------------------------------------------------------------------
// BatchNorm (training-mode): stats over R rows per channel, then apply.
// ---------------------------------------------------------------------------
__global__ void k_bn_stats(const float* __restrict__ X, float* __restrict__ stats,
                           long long R, int C) {
  int c = blockIdx.x;
  float s = 0.f, s2 = 0.f;
  for (long long r = threadIdx.x; r < R; r += blockDim.x) {
    float v = X[r * C + c];
    s += v; s2 += v * v;
  }
  __shared__ float sh[THREADS], sh2[THREADS];
  sh[threadIdx.x] = s; sh2[threadIdx.x] = s2;
  __syncthreads();
  for (int st = THREADS / 2; st > 0; st >>= 1) {
    if ((int)threadIdx.x < st) {
      sh[threadIdx.x]  += sh[threadIdx.x + st];
      sh2[threadIdx.x] += sh2[threadIdx.x + st];
    }
    __syncthreads();
  }
  if (threadIdx.x == 0) {
    float m = sh[0] / (float)R;
    float var = sh2[0] / (float)R - m * m;
    stats[c] = m;
    stats[C + c] = rsqrtf(var + 1e-5f);
  }
}

__global__ void k_bn_apply(const float* __restrict__ X, const float* __restrict__ stats,
                           const float* __restrict__ g, const float* __restrict__ bta,
                           const float* __restrict__ resid, float* __restrict__ Y,
                           long long R, int C, int relu) {
  long long t = (long long)blockIdx.x * blockDim.x + threadIdx.x;
  long long tot = R * C;
  if (t >= tot) return;
  int c = (int)(t % C);
  float v = g[c] * (X[t] - stats[c]) * stats[C + c] + bta[c];
  if (resid) v += resid[t];
  if (relu)  v = fmaxf(v, 0.f);
  Y[t] = v;
}

// ---------------------------------------------------------------------------
// Farthest point sampling: one block per batch, sequential argmax iterations.
// ---------------------------------------------------------------------------
__global__ void k_fps(const float* __restrict__ xyz, int N, int npoint,
                      int* __restrict__ fidx, float* __restrict__ dist) {
  int b = blockIdx.x;
  xyz  += (long long)b * N * 3;
  fidx += (long long)b * npoint;
  dist += (long long)b * N;
  __shared__ float smax[THREADS];
  __shared__ int   simax[THREADS];
  int t = threadIdx.x;
  float p0x = xyz[0], p0y = xyz[1], p0z = xyz[2];
  for (int i = t; i < N; i += blockDim.x) {
    float dx = xyz[i * 3] - p0x, dy = xyz[i * 3 + 1] - p0y, dz = xyz[i * 3 + 2] - p0z;
    dist[i] = dx * dx + dy * dy + dz * dz;
  }
  if (t == 0) fidx[0] = 0;
  __syncthreads();
  for (int it = 1; it < npoint; ++it) {
    float bm = -1.f; int bi = 0;
    for (int i = t; i < N; i += blockDim.x) {
      float d = dist[i];
      if (d > bm) { bm = d; bi = i; }
    }
    smax[t] = bm; simax[t] = bi;
    __syncthreads();
    for (int st = THREADS / 2; st > 0; st >>= 1) {
      if (t < st && smax[t + st] > smax[t]) { smax[t] = smax[t + st]; simax[t] = simax[t + st]; }
      __syncthreads();
    }
    int cur = simax[0];
    if (t == 0) fidx[it] = cur;
    float cx = xyz[cur * 3], cy = xyz[cur * 3 + 1], cz = xyz[cur * 3 + 2];
    for (int i = t; i < N; i += blockDim.x) {
      float dx = xyz[i * 3] - cx, dy = xyz[i * 3 + 1] - cy, dz = xyz[i * 3 + 2] - cz;
      float d = dx * dx + dy * dy + dz * dz;
      if (d < dist[i]) dist[i] = d;
    }
    __syncthreads();
  }
}

// ---------------------------------------------------------------------------
// KNN: one thread per query, insertion top-k (k <= 16).
// ---------------------------------------------------------------------------
__global__ void k_knn(const float* __restrict__ q, const float* __restrict__ ref,
                      int* __restrict__ idx, int B, int M, int N, int k) {
  long long t = (long long)blockIdx.x * blockDim.x + threadIdx.x;
  if (t >= (long long)B * M) return;
  int b = (int)(t / M);
  const float* Q = q + t * 3;
  const float* R = ref + (long long)b * N * 3;
  float qx = Q[0], qy = Q[1], qz = Q[2];
  float bd[16]; int bi[16];
  for (int j = 0; j < k; ++j) { bd[j] = 1e30f; bi[j] = 0; }
  for (int n = 0; n < N; ++n) {
    float dx = qx - R[n * 3], dy = qy - R[n * 3 + 1], dz = qz - R[n * 3 + 2];
    float d = dx * dx + dy * dy + dz * dz;
    if (d < bd[k - 1]) {
      int j = k - 1;
      while (j > 0 && bd[j - 1] > d) { bd[j] = bd[j - 1]; bi[j] = bi[j - 1]; --j; }
      bd[j] = d; bi[j] = n;
    }
  }
  for (int j = 0; j < k; ++j) idx[t * k + j] = bi[j];
}

// out[b,m,j,c] = src[b, idx[b,m,j], c]
__global__ void k_gather(const float* __restrict__ src, const int* __restrict__ idx,
                         float* __restrict__ out, int B, int M, int kk, int N, int C) {
  long long t = (long long)blockIdx.x * blockDim.x + threadIdx.x;
  long long tot = (long long)B * M * kk * C;
  if (t >= tot) return;
  int c = (int)(t % C);
  long long g = t / C;
  long long b = g / ((long long)M * kk);
  int n = idx[g];
  out[t] = src[((long long)b * N + n) * C + c];
}

// rel[b,m,j,:] = src[b, idx] - qp[b,m]
__global__ void k_gather_sub(const float* __restrict__ src, const float* __restrict__ qp,
                             const int* __restrict__ idx, float* __restrict__ out,
                             int B, int M, int kk, int N) {
  long long t = (long long)blockIdx.x * blockDim.x + threadIdx.x;
  long long tot = (long long)B * M * kk;
  if (t >= tot) return;
  long long b = t / ((long long)M * kk);
  long long r = t % ((long long)M * kk);
  long long m = r / kk;
  int n = idx[t];
  const float* s = src + ((long long)b * N + n) * 3;
  const float* Q = qp + ((long long)b * M + m) * 3;
  out[t * 3 + 0] = s[0] - Q[0];
  out[t * 3 + 1] = s[1] - Q[1];
  out[t * 3 + 2] = s[2] - Q[2];
}

// gfeat[b,m,j,:] = [ xyz[idx]-newxyz[b,m] (3), feats[idx] (C) ]
__global__ void k_gfeat(const float* __restrict__ xyz, const float* __restrict__ nxyz,
                        const float* __restrict__ feats, const int* __restrict__ idx,
                        float* __restrict__ out, int B, int M, int kk, int N, int C) {
  long long t = (long long)blockIdx.x * blockDim.x + threadIdx.x;
  long long tot = (long long)B * M * kk;
  if (t >= tot) return;
  long long b = t / ((long long)M * kk);
  long long r = t % ((long long)M * kk);
  long long m = r / kk;
  int n = idx[t];
  const float* s = xyz + ((long long)b * N + n) * 3;
  const float* Q = nxyz + ((long long)b * M + m) * 3;
  float* o = out + t * (3 + C);
  o[0] = s[0] - Q[0]; o[1] = s[1] - Q[1]; o[2] = s[2] - Q[2];
  const float* f = feats + ((long long)b * N + n) * C;
  for (int c = 0; c < C; ++c) o[3 + c] = f[c];
}

// Y[m,c] = max_j X[m,j,c]
__global__ void k_maxk(const float* __restrict__ X, float* __restrict__ Y,
                       long long BM, int k, int C) {
  long long t = (long long)blockIdx.x * blockDim.x + threadIdx.x;
  long long tot = BM * C;
  if (t >= tot) return;
  long long m = t / C; int c = (int)(t % C);
  float mx = -1e30f;
  for (int j = 0; j < k; ++j) mx = fmaxf(mx, X[(m * k + j) * (long long)C + c]);
  Y[t] = mx;
}

// a[n,j,c] = q[n,c] - kf[n,j,c] + pos[n,j,c]   (out may alias kf)
__global__ void k_qkp(const float* __restrict__ q, const float* __restrict__ kf,
                      const float* __restrict__ pos, float* __restrict__ out,
                      long long BN, int k, int C) {
  long long t = (long long)blockIdx.x * blockDim.x + threadIdx.x;
  long long tot = BN * (long long)k * C;
  if (t >= tot) return;
  int c = (int)(t % C);
  long long n = (t / C) / k;
  out[t] = q[n * C + c] - kf[t] + pos[t];
}

__global__ void k_add(const float* __restrict__ a, const float* __restrict__ b,
                      float* __restrict__ o, long long n) {
  long long t = (long long)blockIdx.x * blockDim.x + threadIdx.x;
  if (t < n) o[t] = a[t] + b[t];
}

// softmax over neighbor axis of (BN, k, C)
__global__ void k_softmax_k(const float* __restrict__ X, float* __restrict__ Y,
                            long long BN, int k, int C) {
  long long t = (long long)blockIdx.x * blockDim.x + threadIdx.x;
  long long tot = BN * C;
  if (t >= tot) return;
  long long n = t / C; int c = (int)(t % C);
  const float* x = X + n * (long long)k * C + c;
  float m = -1e30f;
  for (int j = 0; j < k; ++j) m = fmaxf(m, x[(long long)j * C]);
  float s = 0.f;
  for (int j = 0; j < k; ++j) s += expf(x[(long long)j * C] - m);
  float inv = 1.f / s;
  float* y = Y + n * (long long)k * C + c;
  for (int j = 0; j < k; ++j) y[(long long)j * C] = expf(x[(long long)j * C] - m) * inv;
}

// res[n, s*a+ac] = sum_j vv[n,j,s*a+ac] * attn[n,j,ac]
__global__ void k_attn(const float* __restrict__ vv, const float* __restrict__ attn,
                       float* __restrict__ res, long long BN, int k, int C, int a) {
  long long t = (long long)blockIdx.x * blockDim.x + threadIdx.x;
  long long tot = BN * C;
  if (t >= tot) return;
  long long n = t / C; int c = (int)(t % C); int ac = c % a;
  float s = 0.f;
  for (int j = 0; j < k; ++j)
    s += vv[(n * k + j) * (long long)C + c] * attn[(n * k + j) * (long long)a + ac];
  res[t] = s;
}

// prepend cls token, copy feats, extract xyz
__global__ void k_build_input(const float* __restrict__ x, const float* __restrict__ cls,
                              float* __restrict__ feats, float* __restrict__ xyz,
                              int B, int Nin, int C) {
  long long t = (long long)blockIdx.x * blockDim.x + threadIdx.x;
  long long tot = (long long)B * (Nin + 1);
  if (t >= tot) return;
  long long b = t / (Nin + 1);
  long long n = t % (Nin + 1);
  const float* src = (n == 0) ? cls : x + ((long long)b * Nin + (n - 1)) * C;
  float* f = feats + t * C;
  for (int c = 0; c < C; ++c) f[c] = src[c];
  xyz[t * 3 + 0] = src[0]; xyz[t * 3 + 1] = src[1]; xyz[t * 3 + 2] = src[2];
}

__global__ void k_row0(const float* __restrict__ f, float* __restrict__ o,
                       int B, int N, int C) {
  int t = blockIdx.x * blockDim.x + threadIdx.x;
  if (t >= B * C) return;
  int b = t / C, c = t % C;
  o[t] = f[(long long)b * N * C + c];
}

// ===========================================================================
// Host orchestration
// ===========================================================================
namespace {
struct TRp { int Wq,bq,Wk,bk,Wv,bv,dW1,db1,dg,dbt,dW2,db2,gg1,gb1,gW1,gc1,gg2,gb2,gW2,gc2; };
struct BLKp { int W1,g1,b1; TRp tr; int g2,b2,W3,g3,b3; };
struct ENCp { int tdW,tdg,tdb; BLKp blk; };
}

extern "C" void kernel_launch(void* const* d_in, const int* in_sizes, int n_in,
                              void* d_out, int out_size, void* d_ws, size_t ws_size,
                              hipStream_t stream) {
  (void)in_sizes; (void)n_in; (void)out_size; (void)ws_size;
  const int PLANES[5]  = {32, 64, 128, 256, 512};
  const int STRIDES[5] = {1, 4, 4, 4, 4};
  const int NSAMPLE[5] = {8, 16, 16, 16, 16};
  const int B = 8;

  // ---- parameter index map (setup_inputs dict insertion order) ----
  int cnt = 1;                 // 0 = x
  const int cls_token = cnt++; // 1
  ENCp enc[5];
  for (int i = 0; i < 5; ++i) {
    enc[i].tdW = cnt++; enc[i].tdg = cnt++; enc[i].tdb = cnt++;
    BLKp& bl = enc[i].blk;
    bl.W1 = cnt++; bl.g1 = cnt++; bl.b1 = cnt++;
    TRp& t = bl.tr;
    t.Wq = cnt++; t.bq = cnt++; t.Wk = cnt++; t.bk = cnt++; t.Wv = cnt++; t.bv = cnt++;
    t.dW1 = cnt++; t.db1 = cnt++; t.dg = cnt++; t.dbt = cnt++;
    t.dW2 = cnt++; t.db2 = cnt++;
    t.gg1 = cnt++; t.gb1 = cnt++; t.gW1 = cnt++; t.gc1 = cnt++;
    t.gg2 = cnt++; t.gb2 = cnt++; t.gW2 = cnt++; t.gc2 = cnt++;
    bl.g2 = cnt++; bl.b2 = cnt++; bl.W3 = cnt++; bl.g3 = cnt++; bl.b3 = cnt++;
  }
  const int cW1 = cnt++, cb1 = cnt++, cg1 = cnt++, cbt1 = cnt++;
  const int cW2 = cnt++, cb2 = cnt++, cg2 = cnt++, cbt2 = cnt++;
  const int cW3 = cnt++, cb3 = cnt++;

  auto P = [&](int i) -> const float* { return (const float*)d_in[i]; };

  // ---- workspace arena ----
  char* base = (char*)d_ws;
  size_t off = 0;
  auto alloc = [&](long long nfloats) -> float* {
    size_t bytes = ((size_t)nfloats * sizeof(float) + 255) & ~(size_t)255;
    float* p = (float*)(base + off);
    off += bytes;
    return p;
  };
  auto alloci = [&](long long n) -> int* { return (int*)alloc(n); };

  // persistent slots
  float* featsSlot[2] = { alloc(1100000), alloc(1100000) };
  float* xyzSlot[2]   = { alloc(100000),  alloc(100000) };
  half_t* wts = (half_t*)alloc(140000);    // 280k halves >= max Np*Kp (512*512)
  half_t* ats = (half_t*)alloc(8500000);   // 17M halves >= max Bt*Mp*Kp (8*32784*64)
  int fs = 0, xs = 0;
  const size_t mark = off;

  // ---- primitives ----
  auto gemm = [&](const float* A, const float* W, const float* bias, float* C,
                  int Bt, int M, int K, int N, int relu) {
    int Kp = (K + 63) & ~63;
    int Np = (N + 31) & ~31;
    int Mp = (M + 15) & ~15;
    k_cvt_w<<<gblocks((long long)Np * Kp), THREADS, 0, stream>>>(W, wts, K, N, Kp, Np);
    k_cvt_a<<<dim3(gblocks((long long)Mp * Kp), 1, Bt), THREADS, 0, stream>>>(
        A, ats, M, K, Mp, Kp);
    dim3 g(Mp / 16, (Np + 127) / 128, Bt);  // 4 waves x 32 cols per block
    k_gemm<<<g, 128, 0, stream>>>(ats, wts, bias, C, M, Mp, Kp, N, relu);
  };

  auto bn = [&](const float* X, const float* g_, const float* b_, const float* resid,
                float* Y, long long R, int C, int relu) {
    float* stats = alloc(2 * C);
    k_bn_stats<<<C, THREADS, 0, stream>>>(X, stats, R, C);
    k_bn_apply<<<gblocks(R * C), THREADS, 0, stream>>>(X, stats, g_, b_, resid, Y, R, C, relu);
  };

  auto pt_layer = [&](const TRp& t, const float* xyz, const float* f,
                      int Npts, int k, int cp, float* out) {
    long long BN = (long long)B * Npts;
    int a = cp / 8;
    int* idx = alloci(BN * k);
    k_knn<<<gblocks(BN), THREADS, 0, stream>>>(xyz, xyz, idx, B, Npts, Npts, k);
    float* q  = alloc(BN * cp);
    float* kp = alloc(BN * cp);
    float* vp = alloc(BN * cp);
    gemm(f, P(t.Wq), P(t.bq), q,  B, Npts, cp, cp, 0);
    gemm(f, P(t.Wk), P(t.bk), kp, B, Npts, cp, cp, 0);
    gemm(f, P(t.Wv), P(t.bv), vp, B, Npts, cp, cp, 0);
    float* kf = alloc(BN * k * cp);
    float* vg = alloc(BN * k * cp);
    k_gather<<<gblocks(BN * k * cp), THREADS, 0, stream>>>(kp, idx, kf, B, Npts, k, Npts, cp);
    k_gather<<<gblocks(BN * k * cp), THREADS, 0, stream>>>(vp, idx, vg, B, Npts, k, Npts, cp);
    float* rel = alloc(BN * k * 3);
    k_gather_sub<<<gblocks(BN * k), THREADS, 0, stream>>>(xyz, xyz, idx, rel, B, Npts, k, Npts);
    float* h = alloc(BN * k * 3);
    gemm(rel, P(t.dW1), P(t.db1), h, B, Npts * k, 3, 3, 0);
    bn(h, P(t.dg), P(t.dbt), nullptr, h, BN * k, 3, 1);          // in-place
    float* pos = alloc(BN * k * cp);
    gemm(h, P(t.dW2), P(t.db2), pos, B, Npts * k, 3, cp, 0);
    // a = q[:, :, None] - kf + pos   (in-place into kf), then gamma MLP
    k_qkp<<<gblocks(BN * k * cp), THREADS, 0, stream>>>(q, kf, pos, kf, BN, k, cp);
    bn(kf, P(t.gg1), P(t.gb1), nullptr, kf, BN * k, cp, 1);      // in-place
    float* a2 = alloc(BN * k * a);
    gemm(kf, P(t.gW1), P(t.gc1), a2, B, Npts * k, cp, a, 0);
    bn(a2, P(t.gg2), P(t.gb2), nullptr, a2, BN * k, a, 1);       // in-place
    float* a4 = alloc(BN * k * a);
    gemm(a2, P(t.gW2), P(t.gc2), a4, B, Npts * k, a, a, 0);
    float* attn = alloc(BN * k * a);
    k_softmax_k<<<gblocks(BN * a), THREADS, 0, stream>>>(a4, attn, BN, k, a);
    // vv = v + pos (in-place into vg), then shared-head attention reduce
    k_add<<<gblocks(BN * (long long)k * cp), THREADS, 0, stream>>>(
        vg, pos, vg, BN * (long long)k * cp);
    k_attn<<<gblocks(BN * cp), THREADS, 0, stream>>>(vg, attn, out, BN, k, cp, a);
  };

  auto trans_down = [&](int i, const ENCp& e, float*& xyzCur, float*& featsCur,
                        int& Npts, int& Cch) {
    int stride = STRIDES[i], nn = NSAMPLE[i], pl = PLANES[i];
    if (stride == 1) {
      float* tmp = alloc((long long)B * Npts * pl);
      gemm(featsCur, P(e.tdW), nullptr, tmp, B, Npts, Cch, pl, 0);
      float* outF = featsSlot[1 - fs];
      bn(tmp, P(e.tdg), P(e.tdb), nullptr, outF, (long long)B * Npts, pl, 1);
      featsCur = outF; fs = 1 - fs; Cch = pl;
    } else {
      int np = Npts / stride;
      int* fidx = alloci((long long)B * np);
      float* dist = alloc((long long)B * Npts);
      k_fps<<<B, THREADS, 0, stream>>>(xyzCur, Npts, np, fidx, dist);
      float* nxyz = xyzSlot[1 - xs];
      k_gather<<<gblocks((long long)B * np * 3), THREADS, 0, stream>>>(
          xyzCur, fidx, nxyz, B, np, 1, Npts, 3);
      int* idx = alloci((long long)B * np * nn);
      k_knn<<<gblocks((long long)B * np), THREADS, 0, stream>>>(
          nxyz, xyzCur, idx, B, np, Npts, nn);
      int din = 3 + Cch;
      float* gf = alloc((long long)B * np * nn * din);
      k_gfeat<<<gblocks((long long)B * np * nn), THREADS, 0, stream>>>(
          xyzCur, nxyz, featsCur, idx, gf, B, np, nn, Npts, Cch);
      float* h = alloc((long long)B * np * nn * pl);
      gemm(gf, P(e.tdW), nullptr, h, B, np * nn, din, pl, 0);
      bn(h, P(e.tdg), P(e.tdb), nullptr, h, (long long)B * np * nn, pl, 1);  // in-place
      float* outF = featsSlot[1 - fs];
      k_maxk<<<gblocks((long long)B * np * pl), THREADS, 0, stream>>>(
          h, outF, (long long)B * np, nn, pl);
      featsCur = outF; fs = 1 - fs;
      xyzCur = nxyz; xs = 1 - xs;
      Npts = np; Cch = pl;
    }
  };

  auto pt_block = [&](const BLKp& bl, const float* xyz, float* feats,
                      int Npts, int cp, int nsample) -> float* {
    long long BN = (long long)B * Npts;
    float* t1 = alloc(BN * cp);
    gemm(feats, P(bl.W1), nullptr, t1, B, Npts, cp, cp, 0);
    bn(t1, P(bl.g1), P(bl.b1), nullptr, t1, BN, cp, 1);          // in-place
    int k = nsample < Npts ? nsample : Npts;
    float* tl = alloc(BN * cp);
    pt_layer(bl.tr, xyz, t1, Npts, k, cp, tl);
    bn(tl, P(bl.g2), P(bl.b2), nullptr, tl, BN, cp, 1);          // in-place
    float* t3 = alloc(BN * cp);
    gemm(tl, P(bl.W3), nullptr, t3, B, Npts, cp, cp, 0);
    float* outF = featsSlot[1 - fs];
    bn(t3, P(bl.g3), P(bl.b3), feats, outF, BN, cp, 1);          // +identity, ReLU
    fs = 1 - fs;
    return outF;
  };

  // ---- network ----
  int Npts = 4097, Cch = 6;
  k_build_input<<<gblocks((long long)B * Npts), THREADS, 0, stream>>>(
      (const float*)d_in[0], P(cls_token), featsSlot[0], xyzSlot[0], B, 4096, 6);
  float* xyzCur = xyzSlot[0];
  float* featsCur = featsSlot[0];

  for (int i = 0; i < 5; ++i) {
    off = mark;
    trans_down(i, enc[i], xyzCur, featsCur, Npts, Cch);
    off = mark;
    featsCur = pt_block(enc[i].blk, xyzCur, featsCur, Npts, Cch, NSAMPLE[i]);
  }

  // ---- classifier head ----
  off = mark;
  float* crow = alloc(B * 512);
  k_row0<<<gblocks(B * 512), THREADS, 0, stream>>>(featsCur, crow, B, Npts, 512);
  float* h1 = alloc(B * 256);
  gemm(crow, P(cW1), P(cb1), h1, 1, B, 512, 256, 0);
  bn(h1, P(cg1), P(cbt1), nullptr, h1, B, 256, 1);
  float* h2 = alloc(B * 128);
  gemm(h1, P(cW2), P(cb2), h2, 1, B, 256, 128, 0);
  bn(h2, P(cg2), P(cbt2), nullptr, h2, B, 128, 1);
  gemm(h2, P(cW3), P(cb3), (float*)d_out, 1, B, 128, 40, 0);
}
